// BotGCN_12738873000218
// MI455X (gfx1250) — matrix operand order repaired and checked by reading, not compiled, not run on hardware
//
#include <hip/hip_runtime.h>

typedef __attribute__((ext_vector_type(16))) __bf16 v16bf;
typedef __attribute__((ext_vector_type(8)))  float  v8f;

union BFrag { v16bf v; uint4 q[2]; };

__device__ __forceinline__ float lrelu(float x) { return x > 0.0f ? x : 0.01f * x; }

// LDS column strides (bf16 elements) with +8 padding so bank = col*4 + k/2 (mod 64)
#define STRIDE128 136   // 128 + 8 -> 34,816 B for 128 cols
#define STRIDE768 776   // 768 + 8 -> 49,664 B for 32 cols

// ---------------------------------------------------------------------------
// Weight transpose + f32->bf16 convert:  WT[n*K + k] = (bf16) W[k*Ncol + n]
// ---------------------------------------------------------------------------
__global__ void k_transcvt(const float* __restrict__ W, __bf16* __restrict__ WT,
                           int K, int Ncol) {
    int i = blockIdx.x * 256 + threadIdx.x;
    if (i >= K * Ncol) return;
    int k = i % K;
    int n = i / K;
    WT[i] = (__bf16)W[(long)k * Ncol + n];
}

// ---------------------------------------------------------------------------
// Degree helpers
// ---------------------------------------------------------------------------
__global__ void k_fill1(float* __restrict__ p, int n) {
    int i = blockIdx.x * 256 + threadIdx.x;
    if (i < n) p[i] = 1.0f;              // self-loop contributes 1
}
__global__ void k_degcount(const int* __restrict__ dst, float* __restrict__ deg, int E) {
    int e = blockIdx.x * 256 + threadIdx.x;
    if (e < E) atomicAdd(&deg[dst[e]], 1.0f);
}
__global__ void k_rsqrt(float* __restrict__ p, int n) {
    int i = blockIdx.x * 256 + threadIdx.x;
    if (i < n) p[i] = rsqrtf(p[i]);      // deg >= 1 always
}

// ---------------------------------------------------------------------------
// Build bf16 A fragment (16x32 MxK tile) from an f32 row-major matrix.
// Lane l<16 holds row (m0+l), K = {off..off+15}, off=0.
// Lane l>=16 holds row (m0+l-16), off=16.
// ---------------------------------------------------------------------------
__device__ __forceinline__ v16bf load_a_frag(const float* __restrict__ rowptr, int kk, int aoff) {
    const float* ap = rowptr + kk + aoff;
    float4 a0 = *(const float4*)(ap + 0);
    float4 a1 = *(const float4*)(ap + 4);
    float4 a2 = *(const float4*)(ap + 8);
    float4 a3 = *(const float4*)(ap + 12);
    v16bf av;
    av[0]  = (__bf16)a0.x; av[1]  = (__bf16)a0.y; av[2]  = (__bf16)a0.z; av[3]  = (__bf16)a0.w;
    av[4]  = (__bf16)a1.x; av[5]  = (__bf16)a1.y; av[6]  = (__bf16)a1.z; av[7]  = (__bf16)a1.w;
    av[8]  = (__bf16)a2.x; av[9]  = (__bf16)a2.y; av[10] = (__bf16)a2.z; av[11] = (__bf16)a2.w;
    av[12] = (__bf16)a3.x; av[13] = (__bf16)a3.y; av[14] = (__bf16)a3.z; av[15] = (__bf16)a3.w;
    return av;
}

// ---------------------------------------------------------------------------
// Projection GEMM: out[:, colOff:colOff+32] = lrelu(A[N,768] @ W[768,32] + b)
// One wave per 16-row tile; 2 WMMA column tiles; K-loop of 24 x 32.
// WT is [32][768] bf16, staged into padded LDS once per block.
// ---------------------------------------------------------------------------
__global__ void __launch_bounds__(256)
k_proj768(const float* __restrict__ A, const __bf16* __restrict__ WT,
          const float* __restrict__ bias, float* __restrict__ out,
          int nrows, int colOff) {
    __shared__ __bf16 WTs[32 * STRIDE768];
    {   // cooperative stage: 8 threads per column, 192 B (12 uint4) each
        const int col = threadIdx.x >> 3;
        const int seg = threadIdx.x & 7;
        const uint4* gsrc = (const uint4*)(WT + (long)col * 768) + seg * 12;
        uint4*       ldst = (uint4*)(WTs + (long)col * STRIDE768) + seg * 12;
        #pragma unroll
        for (int i = 0; i < 12; ++i) ldst[i] = gsrc[i];
    }
    __syncthreads();

    const int lane = threadIdx.x & 31;
    const int wave = threadIdx.x >> 5;
    const int tile = blockIdx.x * 8 + wave;
    const int m0   = tile * 16;
    if (m0 >= nrows) return;

    const int half = lane >> 4;            // 0 or 1
    const int l15  = lane & 15;
    const int aoff = half * 16;
    const float* rowptr = A + (long)(m0 + l15) * 768;

    v8f acc0 = {}; v8f acc1 = {};
    for (int kk = 0; kk < 768; kk += 32) {
        v16bf av = load_a_frag(rowptr, kk, aoff);
        const int krow = kk + half * 16;   // B: lanes 0-15 K=kk..+15, lanes 16-31 K=kk+16..+31
        BFrag b0, b1;
        const uint4* bp0 = (const uint4*)(WTs + l15 * STRIDE768 + krow);
        b0.q[0] = bp0[0]; b0.q[1] = bp0[1];
        const uint4* bp1 = (const uint4*)(WTs + (16 + l15) * STRIDE768 + krow);
        b1.q[0] = bp1[0]; b1.q[1] = bp1[1];
        acc0 = __builtin_amdgcn_wmma_f32_16x16x32_bf16(false, av, false, b0.v, (short)0, acc0, false, false);
        acc1 = __builtin_amdgcn_wmma_f32_16x16x32_bf16(false, av, false, b1.v, (short)0, acc1, false, false);
    }
    const int rbase = m0 + half * 8;       // C layout: VGPR r -> M=r (lanes 0-15), M=r+8 (16-31)
    const float bv0 = bias[l15];
    const float bv1 = bias[16 + l15];
    #pragma unroll
    for (int r = 0; r < 8; ++r) {
        out[(long)(rbase + r) * 128 + colOff + l15]      = lrelu(acc0[r] + bv0);
        out[(long)(rbase + r) * 128 + colOff + 16 + l15] = lrelu(acc1[r] + bv1);
    }
}

// ---------------------------------------------------------------------------
// 128x128 GEMM: out = (lrelu?)(A[N,128] @ W[128,128] (+ bias))
// Wave computes 16 rows x 128 cols: 8 accumulators, 4 K-iters x 8 WMMAs.
// WT staged into padded LDS once per block.
// ---------------------------------------------------------------------------
__global__ void __launch_bounds__(256)
k_gemm128(const float* __restrict__ A, const __bf16* __restrict__ WT,
          const float* __restrict__ bias, float* __restrict__ out,
          int nrows, int doLrelu) {
    __shared__ __bf16 WTs[128 * STRIDE128];
    {   // cooperative stage: 2 threads per column, 64 B (4 uint4) each
        const int col  = threadIdx.x >> 1;
        const int hseg = threadIdx.x & 1;
        const uint4* gsrc = (const uint4*)(WT + (long)col * 128) + hseg * 4;
        uint4*       ldst = (uint4*)(WTs + (long)col * STRIDE128) + hseg * 4;
        #pragma unroll
        for (int i = 0; i < 4; ++i) ldst[i] = gsrc[i];
    }
    __syncthreads();

    const int lane = threadIdx.x & 31;
    const int wave = threadIdx.x >> 5;
    const int tile = blockIdx.x * 8 + wave;
    const int m0   = tile * 16;
    if (m0 >= nrows) return;

    const int half = lane >> 4;
    const int l15  = lane & 15;
    const int aoff = half * 16;
    const float* rowptr = A + (long)(m0 + l15) * 128;

    v8f acc[8] = {};
    for (int kk = 0; kk < 128; kk += 32) {
        v16bf av = load_a_frag(rowptr, kk, aoff);
        const int krow = kk + half * 16;
        #pragma unroll
        for (int c = 0; c < 8; ++c) {
            BFrag b;
            const uint4* bp = (const uint4*)(WTs + (c * 16 + l15) * STRIDE128 + krow);
            b.q[0] = bp[0]; b.q[1] = bp[1];
            acc[c] = __builtin_amdgcn_wmma_f32_16x16x32_bf16(false, av, false, b.v, (short)0, acc[c], false, false);
        }
    }
    const int rbase = m0 + half * 8;
    #pragma unroll
    for (int c = 0; c < 8; ++c) {
        const int col = c * 16 + l15;
        const float bv = bias ? bias[col] : 0.0f;
        #pragma unroll
        for (int r = 0; r < 8; ++r) {
            float v = acc[c][r] + bv;
            out[(long)(rbase + r) * 128 + col] = doLrelu ? lrelu(v) : v;
        }
    }
}

// ---------------------------------------------------------------------------
// Small projections: cols 64..95 = lrelu(num[5] @ Wn[5,32] + bn)
//                    cols 96..127 = lrelu(cat[1] * Wc[1,32] + bc)
// ---------------------------------------------------------------------------
__global__ void k_smallproj(const float* __restrict__ num, const float* __restrict__ cat,
                            const float* __restrict__ Wn, const float* __restrict__ bn,
                            const float* __restrict__ Wc, const float* __restrict__ bc,
                            float* __restrict__ out, int N) {
    int idx = blockIdx.x * 256 + threadIdx.x;
    if (idx >= N * 64) return;
    int row = idx >> 6;
    int c   = idx & 63;
    float v; int col;
    if (c < 32) {
        float s = bn[c];
        #pragma unroll
        for (int k = 0; k < 5; ++k) s += num[(long)row * 5 + k] * Wn[k * 32 + c];
        v = s; col = 64 + c;
    } else {
        int cc = c - 32;
        v = cat[row] * Wc[cc] + bc[cc];
        col = 96 + cc;
    }
    out[(long)row * 128 + col] = lrelu(v);
}

// ---------------------------------------------------------------------------
// GCN self-loop init: out[i,f] = h[i,f]*dinv[i]^2 + bias[f]
// ---------------------------------------------------------------------------
__global__ void k_gcn_self(const float* __restrict__ h, const float* __restrict__ dinv,
                           const float* __restrict__ bias, float* __restrict__ out, int N) {
    long idx = (long)blockIdx.x * 256 + threadIdx.x;
    if (idx >= (long)N * 128) return;
    int i = (int)(idx >> 7);
    int f = (int)(idx & 127);
    float di = dinv[i];
    out[idx] = h[idx] * di * di + bias[f];
}

// ---------------------------------------------------------------------------
// GCN edge scatter, warp-cooperative batches of 32 edges.
// Phase 1 (coalesced): lane e loads (src,dst) and computes norm = dinv[s]*dinv[d].
// Phase 2: broadcast edge i via shuffles; lane owns a float4 of the 128 feats.
// out[dst] += h[src] * norm   (exact segment-sum semantics via f32 atomics)
// ---------------------------------------------------------------------------
__global__ void __launch_bounds__(256)
k_gcn_edge(const float* __restrict__ h, const float* __restrict__ dinv,
           const int* __restrict__ src, const int* __restrict__ dst,
           float* __restrict__ out, int E) {
    const int lane  = threadIdx.x & 31;
    const int wave  = threadIdx.x >> 5;
    const long ebase = ((long)blockIdx.x * 8 + wave) * 32;
    if (ebase >= E) return;

    int s = 0, d = 0;
    float nrm = 0.0f;
    const long e = ebase + lane;
    if (e < E) {
        s = src[e];
        d = dst[e];
        nrm = dinv[s] * dinv[d];
    }
    const int nE = (int)((E - ebase) < 32 ? (E - ebase) : 32);
    const int f  = lane * 4;

    for (int i = 0; i < nE; ++i) {
        const int   si = __shfl(s, i, 32);
        const int   di = __shfl(d, i, 32);
        const float ni = __shfl(nrm, i, 32);
        float4 hv = *(const float4*)(h + (long)si * 128 + f);
        float* op = out + (long)di * 128 + f;
        atomicAdd(op + 0, hv.x * ni);
        atomicAdd(op + 1, hv.y * ni);
        atomicAdd(op + 2, hv.z * ni);
        atomicAdd(op + 3, hv.w * ni);
    }
}

// ---------------------------------------------------------------------------
// Final projection: out[N,2] = x[N,128] @ W[128,2] + b
// ---------------------------------------------------------------------------
__global__ void __launch_bounds__(128)
k_out2(const float* __restrict__ x, const float* __restrict__ W,
       const float* __restrict__ b, float* __restrict__ out, int N) {
    __shared__ float Ws[256];
    Ws[threadIdx.x]       = W[threadIdx.x];
    Ws[threadIdx.x + 128] = W[threadIdx.x + 128];
    __syncthreads();
    int node = blockIdx.x * 128 + threadIdx.x;
    if (node >= N) return;
    float a0 = b[0], a1 = b[1];
    const float* xr = x + (long)node * 128;
    #pragma unroll 8
    for (int k = 0; k < 128; k += 4) {
        float4 xv = *(const float4*)(xr + k);
        a0 += xv.x * Ws[(k + 0) * 2]     + xv.y * Ws[(k + 1) * 2]
            + xv.z * Ws[(k + 2) * 2]     + xv.w * Ws[(k + 3) * 2];
        a1 += xv.x * Ws[(k + 0) * 2 + 1] + xv.y * Ws[(k + 1) * 2 + 1]
            + xv.z * Ws[(k + 2) * 2 + 1] + xv.w * Ws[(k + 3) * 2 + 1];
    }
    out[(long)node * 2 + 0] = a0;
    out[(long)node * 2 + 1] = a1;
}

// ---------------------------------------------------------------------------
extern "C" void kernel_launch(void* const* d_in, const int* in_sizes, int n_in,
                              void* d_out, int out_size, void* d_ws, size_t ws_size,
                              hipStream_t stream) {
    const float* des   = (const float*)d_in[0];
    const float* tweet = (const float*)d_in[1];
    const float* num   = (const float*)d_in[2];
    const float* cat   = (const float*)d_in[3];
    const int*   ei    = (const int*)d_in[4];
    const float* W_des = (const float*)d_in[5];  const float* b_des = (const float*)d_in[6];
    const float* W_tw  = (const float*)d_in[7];  const float* b_tw  = (const float*)d_in[8];
    const float* W_num = (const float*)d_in[9];  const float* b_num = (const float*)d_in[10];
    const float* W_cat = (const float*)d_in[11]; const float* b_cat = (const float*)d_in[12];
    const float* W_in  = (const float*)d_in[13]; const float* b_in  = (const float*)d_in[14];
    const float* W_g1  = (const float*)d_in[15]; const float* b_g1  = (const float*)d_in[16];
    const float* W_g2  = (const float*)d_in[17]; const float* b_g2  = (const float*)d_in[18];
    const float* W_o1  = (const float*)d_in[19]; const float* b_o1  = (const float*)d_in[20];
    const float* W_o2  = (const float*)d_in[21]; const float* b_o2  = (const float*)d_in[22];

    const int N = in_sizes[0] / 768;
    const int E = in_sizes[4] / 2;
    const int* srcp = ei;
    const int* dstp = ei + E;

    // Workspace carve-up
    char* ws = (char*)d_ws;
    size_t off = 0;
    auto carve = [&](size_t bytes) { void* p = ws + off; off = (off + bytes + 255) & ~(size_t)255; return p; };
    float*  bufA    = (float*)carve((size_t)N * 128 * sizeof(float));
    float*  bufB    = (float*)carve((size_t)N * 128 * sizeof(float));
    float*  dinv    = (float*)carve((size_t)N * sizeof(float));
    __bf16* wt_des  = (__bf16*)carve((size_t)768 * 32 * sizeof(__bf16));
    __bf16* wt_tw   = (__bf16*)carve((size_t)768 * 32 * sizeof(__bf16));
    __bf16* wt_in   = (__bf16*)carve((size_t)128 * 128 * sizeof(__bf16));
    __bf16* wt_g1   = (__bf16*)carve((size_t)128 * 128 * sizeof(__bf16));
    __bf16* wt_g2   = (__bf16*)carve((size_t)128 * 128 * sizeof(__bf16));
    __bf16* wt_o1   = (__bf16*)carve((size_t)128 * 128 * sizeof(__bf16));

    const int T = 256;
    // Weights -> bf16 transposed
    k_transcvt<<<(768 * 32 + T - 1) / T, T, 0, stream>>>(W_des, wt_des, 768, 32);
    k_transcvt<<<(768 * 32 + T - 1) / T, T, 0, stream>>>(W_tw,  wt_tw,  768, 32);
    k_transcvt<<<(128 * 128 + T - 1) / T, T, 0, stream>>>(W_in, wt_in, 128, 128);
    k_transcvt<<<(128 * 128 + T - 1) / T, T, 0, stream>>>(W_g1, wt_g1, 128, 128);
    k_transcvt<<<(128 * 128 + T - 1) / T, T, 0, stream>>>(W_g2, wt_g2, 128, 128);
    k_transcvt<<<(128 * 128 + T - 1) / T, T, 0, stream>>>(W_o1, wt_o1, 128, 128);

    // Degrees -> dinv
    k_fill1<<<(N + T - 1) / T, T, 0, stream>>>(dinv, N);
    k_degcount<<<(E + T - 1) / T, T, 0, stream>>>(dstp, dinv, E);
    k_rsqrt<<<(N + T - 1) / T, T, 0, stream>>>(dinv, N);

    const int tiles  = (N + 15) / 16;
    const int blocks = (tiles + 7) / 8;

    // Feature projections -> bufA [N,128]
    k_proj768<<<blocks, T, 0, stream>>>(des,   wt_des, b_des, bufA, N, 0);
    k_proj768<<<blocks, T, 0, stream>>>(tweet, wt_tw,  b_tw,  bufA, N, 32);
    k_smallproj<<<((long)N * 64 + T - 1) / T, T, 0, stream>>>(num, cat, W_num, b_num, W_cat, b_cat, bufA, N);

    // x1 = lrelu(x0 @ W_in + b_in) -> bufB
    k_gemm128<<<blocks, T, 0, stream>>>(bufA, wt_in, b_in, bufB, N, 1);

    const long edgeBlocks = ((long)E + 255) / 256;   // 8 waves/block, 32 edges/wave
    const long nfBlocks   = ((long)N * 128 + T - 1) / T;

    // GCN conv 1: h = x1 @ W_g1 -> bufA ; out = self + scatter + b -> bufB
    k_gemm128<<<blocks, T, 0, stream>>>(bufB, wt_g1, nullptr, bufA, N, 0);
    k_gcn_self<<<nfBlocks, T, 0, stream>>>(bufA, dinv, b_g1, bufB, N);
    k_gcn_edge<<<edgeBlocks, T, 0, stream>>>(bufA, dinv, srcp, dstp, bufB, E);

    // GCN conv 2
    k_gemm128<<<blocks, T, 0, stream>>>(bufB, wt_g2, nullptr, bufA, N, 0);
    k_gcn_self<<<nfBlocks, T, 0, stream>>>(bufA, dinv, b_g2, bufB, N);
    k_gcn_edge<<<edgeBlocks, T, 0, stream>>>(bufA, dinv, srcp, dstp, bufB, E);

    // x3 = lrelu(x @ W_o1 + b_o1) -> bufA
    k_gemm128<<<blocks, T, 0, stream>>>(bufB, wt_o1, b_o1, bufA, N, 1);

    // out = x3 @ W_o2 + b_o2
    k_out2<<<(N + 127) / 128, 128, 0, stream>>>(bufA, W_o2, b_o2, (float*)d_out, N);
}
